// ATSBlock_57732950393451
// MI455X (gfx1250) — matrix operand order repaired and checked by reading, not compiled, not run on hardware
//
#include <hip/hip_runtime.h>
#include <hip/hip_bf16.h>
#include <math.h>

typedef _Float16 half_t;
typedef __attribute__((ext_vector_type(16))) _Float16 v16h;
typedef __attribute__((ext_vector_type(8)))  _Float16 v8h;
typedef __attribute__((ext_vector_type(8)))  float    v8f;

constexpr int BQ    = 8;
constexpr int NQ    = 785;
constexpr int NPADQ = 800;  // padded attention row stride (16B-aligned, mult of 32)
constexpr int ATTNR = 786;  // attention rows per (b,h): 785 real + 1 zero dummy
constexpr int KVPAD = 832;  // padded K/V token rows per (b,h) (13*64)
constexpr int CQ    = 768;
constexpr int HQ    = 12;
constexpr int DHQ   = 64;
constexpr int NTQ   = 393;  // n_tokens
constexpr int NT1Q  = 392;  // n_tokens - 1
constexpr int NM1Q  = 784;  // N - 1
constexpr int NREFQ = 785;  // n_ref_tokens

// ---------------------------------------------------------------- utilities
__global__ __launch_bounds__(256) void zero_kernel(float* a, long n) {
  long i = (long)blockIdx.x * 256 + threadIdx.x;
  for (; i < n; i += (long)gridDim.x * 256) a[i] = 0.f;
}

__global__ __launch_bounds__(256) void zero_half_kernel(half_t* a, long n) {
  long i = (long)blockIdx.x * 256 + threadIdx.x;
  for (; i < n; i += (long)gridDim.x * 256) a[i] = (half_t)0.f;
}

// zero the dummy attention row (index NQ) of each (b,h) slice
__global__ __launch_bounds__(256) void zero_attn_dummy_kernel(half_t* attn) {
  long z = blockIdx.x;
  half_t* r = attn + (z * ATTNR + NQ) * (long)NPADQ;
  for (int c = threadIdx.x; c < NPADQ; c += 256) r[c] = (half_t)0.f;
}

__global__ __launch_bounds__(256) void cvt_kernel(const float* __restrict__ s,
                                                  half_t* __restrict__ d, long n) {
  long i = (long)blockIdx.x * 256 + threadIdx.x;
  for (; i < n; i += (long)gridDim.x * 256) d[i] = (half_t)s[i];
}

// ---------------------------------------------------------------- layernorm
__global__ __launch_bounds__(256) void ln_kernel(const float* __restrict__ x,
                                                 const float* __restrict__ w,
                                                 const float* __restrict__ bb,
                                                 half_t* __restrict__ out) {
  __shared__ float red[256];
  long row = blockIdx.x;
  const float* xr = x + row * CQ;
  int t = threadIdx.x;
  float s = 0.f;
  for (int c = t; c < CQ; c += 256) s += xr[c];
  red[t] = s; __syncthreads();
  for (int o = 128; o > 0; o >>= 1) { if (t < o) red[t] += red[t + o]; __syncthreads(); }
  float mean = red[0] / (float)CQ; __syncthreads();
  float v = 0.f;
  for (int c = t; c < CQ; c += 256) { float d = xr[c] - mean; v += d * d; }
  red[t] = v; __syncthreads();
  for (int o = 128; o > 0; o >>= 1) { if (t < o) red[t] += red[t + o]; __syncthreads(); }
  float rstd = rsqrtf(red[0] / (float)CQ + 1e-5f);
  for (int c = t; c < CQ; c += 256)
    out[row * CQ + c] = (half_t)((xr[c] - mean) * rstd * w[c] + bb[c]);
}

// --------------------------------------------------- softmax with policy (+sig)
__global__ __launch_bounds__(256) void softmax_kernel(half_t* __restrict__ attn,
                                                      const float* __restrict__ policy,
                                                      float* __restrict__ sigraw) {
  __shared__ float red[256];
  long gid = blockIdx.x;
  int row = (int)(gid % NQ);
  long z = gid / NQ;
  int b = (int)(z / HQ);
  half_t* ar = attn + (z * ATTNR + row) * (long)NPADQ;
  int t = threadIdx.x;
  float m = -1e30f;
  for (int c = t; c < NQ; c += 256) m = fmaxf(m, (float)ar[c]);
  red[t] = m; __syncthreads();
  for (int o = 128; o > 0; o >>= 1) { if (t < o) red[t] = fmaxf(red[t], red[t + o]); __syncthreads(); }
  m = red[0]; __syncthreads();
  float s = 0.f;
  for (int c = t; c < NQ; c += 256) {
    float p = policy[(long)b * NQ + c];
    if (c == row) p = p + (1.f - p);
    float e = __expf((float)ar[c] - m) * p;
    ar[c] = (half_t)e;
    s += e;
  }
  red[t] = s; __syncthreads();
  for (int o = 128; o > 0; o >>= 1) { if (t < o) red[t] += red[t + o]; __syncthreads(); }
  s = red[0];
  float inv = 1.f / (s + 1e-6f);
  const float addc = 1e-6f / (float)NQ;
  for (int c = t; c < NQ; c += 256) {
    float o2 = ((float)ar[c] + addc) * inv;
    ar[c] = (half_t)o2;
    if (row == 0) atomicAdd(&sigraw[(long)b * NQ + c], o2);
  }
}

// ---------------------------------------------------------------- significance
__global__ __launch_bounds__(1024) void sig_kernel(const float* __restrict__ sigraw,
                                                   const float* __restrict__ vn2,
                                                   float* __restrict__ sig) {
  __shared__ float red[1024];
  int b = blockIdx.x, t = threadIdx.x;
  float v = 0.f;
  if (t < NM1Q) v = sigraw[(long)b * NQ + 1 + t] * sqrtf(vn2[(long)b * NQ + 1 + t]);
  red[t] = v; __syncthreads();
  for (int o = 512; o > 0; o >>= 1) { if (t < o) red[t] += red[t + o]; __syncthreads(); }
  float s = red[0];
  if (t < NM1Q) sig[(long)b * NM1Q + t] = v / s;
}

// --------------------------------- sort / CDF / token-pick / unique (per batch)
__global__ __launch_bounds__(1024) void select_kernel(const float* __restrict__ sig,
                                                      int* __restrict__ rowsrc,
                                                      float* __restrict__ pol_out) {
  __shared__ float key[1024];
  __shared__ int   idx[1024];
  __shared__ float buf[1024];
  __shared__ int   ordS[1024];
  __shared__ float s_cmin, s_cmax, s_ys;
  int b = blockIdx.x, t = threadIdx.x;
  key[t] = (t < NM1Q) ? sig[(long)b * NM1Q + t] : 1e30f;
  idx[t] = t;
  __syncthreads();
  // bitonic ascending on (value, index) — stable-argsort equivalent
  for (int k = 2; k <= 1024; k <<= 1) {
    for (int j = k >> 1; j > 0; j >>= 1) {
      int p = t ^ j;
      if (p > t) {
        float a = key[t], c2 = key[p];
        int ia = idx[t], ic = idx[p];
        bool up = ((t & k) == 0);
        bool gt = (a > c2) || ((a == c2) && (ia > ic));
        if (gt == up) { key[t] = c2; key[p] = a; idx[t] = ic; idx[p] = ia; }
      }
      __syncthreads();
    }
  }
  ordS[t] = idx[t];
  __syncthreads();
  // inclusive scan of sorted scores -> cdf
  float xv = (t < NM1Q) ? key[t] : 0.f;
  buf[t] = xv; __syncthreads();
  for (int off = 1; off < 1024; off <<= 1) {
    float add = (t >= off) ? buf[t - off] : 0.f;
    __syncthreads();
    buf[t] += add;
    __syncthreads();
  }
  if (t == 0) { s_cmin = buf[0]; s_cmax = buf[NM1Q - 1]; }
  __syncthreads();
  float ncdf = (t < NM1Q) ? (buf[t] - s_cmin) / (s_cmax - s_cmin) : 1e30f;
  key[t] = ncdf;
  __syncthreads();
  float mv = (t < NM1Q) ? (ncdf + (ncdf == 0.f ? 1e8f : 0.f)) : 1e30f;
  buf[t] = mv; __syncthreads();
  for (int o = 512; o > 0; o >>= 1) { if (t < o) buf[t] = fminf(buf[t], buf[t + o]); __syncthreads(); }
  if (t == 0) s_ys = buf[0];
  __syncthreads();
  // tokens_to_pick
  const int diff = (NREFQ - 1) - (NQ - 1);
  int tok = 0x7fffffff;
  if (t < NREFQ - 1) {
    float ysi = s_ys + (float)t * (1.f - s_ys) / (float)(NREFQ - 2);
    float best = 1e30f; int bj = 0;
    for (int jj = 0; jj < NREFQ - 1; jj++) {
      float nv = (jj < diff) ? 0.f : key[jj - diff];
      float d = fabsf(ysi - nv);
      if (d < best) { best = d; bj = jj; }
    }
    tok = bj - diff;
  }
  __syncthreads();
  idx[t] = tok;
  __syncthreads();
  // bitonic int sort ascending
  for (int k = 2; k <= 1024; k <<= 1) {
    for (int j = k >> 1; j > 0; j >>= 1) {
      int p = t ^ j;
      if (p > t) {
        int a = idx[t], c2 = idx[p];
        bool up = ((t & k) == 0);
        if ((a > c2) == up) { idx[t] = c2; idx[p] = a; }
      }
      __syncthreads();
    }
  }
  // dedupe: equal-to-next -> max_value(N-1)
  int sv = idx[t];
  int sh = (t == NM1Q - 1) ? 1 : ((t < NM1Q - 1) ? idx[t + 1] : sv);
  int uv = (t < NM1Q) ? ((sh == sv) ? NM1Q : sv) : 0x7fffffff;
  __syncthreads();
  idx[t] = uv;
  __syncthreads();
  for (int k = 2; k <= 1024; k <<= 1) {
    for (int j = k >> 1; j > 0; j >>= 1) {
      int p = t ^ j;
      if (p > t) {
        int a = idx[t], c2 = idx[p];
        bool up = ((t & k) == 0);
        if ((a > c2) == up) { idx[t] = c2; idx[p] = a; }
      }
      __syncthreads();
    }
  }
  if (t < NT1Q) {
    int u = idx[t];
    bool valid = (u != NM1Q);
    rowsrc[(long)b * NTQ + 1 + t] = valid ? (1 + ordS[u]) : -1;
    pol_out[(long)b * NTQ + 1 + t] = valid ? 1.f : 0.f;
  }
  if (t == 0) { rowsrc[(long)b * NTQ] = 0; pol_out[(long)b * NTQ] = 1.f; }
}

// ------------------------------------------------------------ gather selected_x
__global__ __launch_bounds__(256) void selx_kernel(const float* __restrict__ x,
                                                   const int* __restrict__ rowsrc,
                                                   float* __restrict__ x2) {
  int gid = blockIdx.x;
  int b = gid / NTQ, m = gid % NTQ;
  int rs = rowsrc[(long)b * NTQ + m];
  float* dst = x2 + ((long)b * NTQ + m) * CQ;
  const float* src = x + ((long)b * NQ + (rs >= 0 ? rs : 0)) * CQ;
  for (int c = threadIdx.x; c < CQ; c += 256) dst[c] = (rs >= 0) ? src[c] : 0.f;
}

// --------------------------------------------------------- generic WMMA GEMM
// C(M,Nn) = A(M,K) x W(K,Nn), f16 in, f32 accum, templated epilogue.
// Double-buffered LDS staging (stage k+1 under compute of k), branch-free
// staging via launcher-enforced padding:
//  - K is a multiple of 32; all staged A/B segments in-bounds, 16B-aligned.
//  - Out-of-range A rows clamped (masked in epilogue).
//  - amap (row indirection) maps invalid rows to a zeroed dummy row (zrow).
struct GemmP {
  const half_t* A; long aStride; int lda;
  const half_t* W; long wStride; int ldw; int wtrans;  // wtrans: Welem(k,n)=W[n*ldw+k]
  int M, Nn, K;
  const int* amap; int amapDiv; int zrow;
  float* outF;
  half_t* outH; long oHStride; int oHld;
  const float* bias;
  const float* rowScale;
  float alpha;
  float* vnorm2;
  half_t *qh, *kh, *vh;
};

// EPI: 1=QKV split, 2=f16 scaled scores, 3=AV->out_h, 4=proj accum, 5=fc1 gelu, 6=fc2 accum
template <int EPI>
__global__ __launch_bounds__(128) void gemm_kernel(GemmP p) {
  __shared__ __attribute__((aligned(16))) half_t Alds[2][64 * 40];
  __shared__ __attribute__((aligned(16))) half_t Blds[2][64 * 40];
  int z = blockIdx.z;
  int rowBase = blockIdx.x * 64, colBase = blockIdx.y * 64;
  int tid = threadIdx.x, lane = tid & 31, wave = tid >> 5;
  const half_t* Ab = p.A + (long)z * p.aStride;
  const half_t* Wb = p.W + (long)z * p.wStride;
  v8f acc[4] = {};

  // A staging geometry (K-invariant): thread t -> row t/2, half-block t&1
  const int ar_ = tid >> 1, ac0 = (tid & 1) * 16;
  int agr = rowBase + ar_;
  int arow = (agr < p.M) ? agr : (p.M - 1);           // clamp: always readable
  if (p.amap) {
    int mrow = p.amap[((long)(z / p.amapDiv)) * p.M + arow];
    arow = (mrow < 0) ? p.zrow : mrow;                // invalid -> zero dummy row
  }
  const half_t* aRowPtr = Ab + (long)arow * p.lda + ac0;
  // B staging geometry
  const int bk_ = tid >> 2, bn0 = (tid & 3) * 16;       // !wtrans
  const int tn_ = tid >> 1, tc0 = (tid & 1) * 16;       // wtrans
  const half_t* bRowPtr0 = Wb + (long)bk_ * p.ldw + colBase + bn0;  // !wtrans (+= k*ldw)
  const half_t* bRowPtrT = Wb + (long)(colBase + tn_) * p.ldw + tc0; // wtrans (+= k)

  auto stage = [&](int k0, int bi) {
    { // A: 64x32 halves
      const half_t* src = aRowPtr + k0;
      *(v8h*)&Alds[bi][ar_ * 40 + ac0]     = *(const v8h*)(src);
      *(v8h*)&Alds[bi][ar_ * 40 + ac0 + 8] = *(const v8h*)(src + 8);
    }
    if (!p.wtrans) { // B 32(k)x64(n): store K-contiguous (transposed)
      const half_t* src = bRowPtr0 + (long)k0 * p.ldw;
      v8h t0 = *(const v8h*)(src);
      v8h t1 = *(const v8h*)(src + 8);
#pragma unroll
      for (int i = 0; i < 8; i++) Blds[bi][(bn0 + i) * 40 + bk_] = t0[i];
#pragma unroll
      for (int i = 0; i < 8; i++) Blds[bi][(bn0 + 8 + i) * 40 + bk_] = t1[i];
    } else { // W is (Nn x K) row-major: already K-contiguous
      const half_t* src = bRowPtrT + k0;
      *(v8h*)&Blds[bi][tn_ * 40 + tc0]     = *(const v8h*)(src);
      *(v8h*)&Blds[bi][tn_ * 40 + tc0 + 8] = *(const v8h*)(src + 8);
    }
  };

  const int nk = p.K >> 5;   // K / 32 tiles
  stage(0, 0);
  __syncthreads();

  const int mrow = wave * 16 + (lane & 15);
  const int kb = (lane < 16) ? 0 : 8;
  const int khalf = (lane < 16) ? 0 : 16;

  for (int it = 0; it < nk; it++) {
    int cur = it & 1;
    if (it + 1 < nk) {
      // prefetch tile it+2 toward the WGP while the pipe runs
      if (it + 2 < nk) {
        int kpf = (it + 2) << 5;
        __builtin_prefetch((const void*)(aRowPtr + kpf), 0, 3);
        if (!p.wtrans) __builtin_prefetch((const void*)(bRowPtr0 + (long)kpf * p.ldw), 0, 3);
        else           __builtin_prefetch((const void*)(bRowPtrT + kpf), 0, 3);
      }
      stage((it + 1) << 5, cur ^ 1);   // fill other buffer under compute
    }
    // compute from buffer `cur`
    v8h a0 = *(const v8h*)&Alds[cur][mrow * 40 + kb];
    v8h a1 = *(const v8h*)&Alds[cur][mrow * 40 + kb + 16];
    v16h a;
#pragma unroll
    for (int i = 0; i < 8; i++) { a[i] = a0[i]; a[i + 8] = a1[i]; }
#pragma unroll
    for (int j = 0; j < 4; j++) {
      int ncol = j * 16 + (lane & 15);
      v8h b0 = *(const v8h*)&Blds[cur][ncol * 40 + kb];
      v8h b1 = *(const v8h*)&Blds[cur][ncol * 40 + kb + 16];
      v16h bf;
#pragma unroll
      for (int i = 0; i < 8; i++) { bf[i] = b0[i]; bf[i + 8] = b1[i]; }
      acc[j] = __builtin_amdgcn_wmma_f32_16x16x32_f16(
          false, a, false, bf, (short)0, acc[j], false, false);
    }
    __syncthreads();
  }

  // epilogue
#pragma unroll
  for (int j = 0; j < 4; j++) {
#pragma unroll
    for (int r = 0; r < 8; r++) {
      int row = rowBase + wave * 16 + ((lane < 16) ? r : r + 8);
      int col = colBase + j * 16 + (lane & 15);
      if (row >= p.M || col >= p.Nn) continue;
      float val = acc[j][r];
      if constexpr (EPI == 1) {  // QKV: policy scale, split, ||v||^2
        int b = z;
        float pol = p.rowScale[(long)b * NQ + row];
        val *= pol;
        int s = col / CQ;
        int h = (col % CQ) / DHQ;
        int d = col % DHQ;
        if (s == 0) {
          p.qh[(((long)b * HQ + h) * NQ + row) * DHQ + d] = (half_t)val;
        } else if (s == 1) {
          p.kh[(((long)b * HQ + h) * KVPAD + row) * DHQ + d] = (half_t)val;
        } else {
          p.vh[(((long)b * HQ + h) * KVPAD + row) * DHQ + d] = (half_t)val;
          atomicAdd(&p.vnorm2[(long)b * NQ + row], val * val);
        }
      } else if constexpr (EPI == 2) {  // scaled attention scores (f16, padded ld)
        p.outH[(long)z * p.oHStride + (long)row * p.oHld + col] = (half_t)(val * p.alpha);
      } else if constexpr (EPI == 3) {  // AV -> [b, token, h*64+d] f16
        int b = z / HQ, h = z % HQ;
        p.outH[((long)b * NTQ + row) * CQ + h * DHQ + col] = (half_t)val;
      } else if constexpr (EPI == 4 || EPI == 6) {  // bias + pol_new, accumulate
        float pol = p.rowScale[(long)z * NTQ + row];
        long o = ((long)z * NTQ + row) * CQ + col;
        p.outF[o] += (val + p.bias[col]) * pol;
      } else if constexpr (EPI == 5) {  // fc1 bias + exact GELU -> f16
        float u = val + p.bias[col];
        float g = 0.5f * u * (1.f + erff(u * 0.70710678118f));
        p.outH[((long)z * NTQ + row) * (4 * CQ) + col] = (half_t)g;
      } else {
        p.outF[(long)z * (long)p.M * p.Nn + (long)row * p.Nn + col] = val;
      }
    }
  }
}

// ------------------------------------------------------------------- launcher
extern "C" void kernel_launch(void* const* d_in, const int* in_sizes, int n_in,
                              void* d_out, int out_size, void* d_ws, size_t ws_size,
                              hipStream_t stream) {
  const float* x     = (const float*)d_in[0];
  const float* polin = (const float*)d_in[1];
  const float* n1w   = (const float*)d_in[2];
  const float* n1b   = (const float*)d_in[3];
  const float* qkvw  = (const float*)d_in[4];
  const float* projw = (const float*)d_in[5];
  const float* projb = (const float*)d_in[6];
  const float* n2w   = (const float*)d_in[7];
  const float* n2b   = (const float*)d_in[8];
  const float* fc1w  = (const float*)d_in[9];
  const float* fc1b  = (const float*)d_in[10];
  const float* fc2w  = (const float*)d_in[11];
  const float* fc2b  = (const float*)d_in[12];

  float* x2  = (float*)d_out;                              // [B, NT, C]
  float* pol = (float*)d_out + (size_t)BQ * NTQ * CQ;      // [B, NT]

  char* ws = (char*)d_ws;
  size_t off = 0;
  auto take = [&](size_t bytes) -> char* {
    char* p = ws + off;
    off = (off + bytes + 255) & ~(size_t)255;
    return p;
  };
  half_t* xnh    = (half_t*)take(sizeof(half_t) * (size_t)BQ * NQ * CQ);
  half_t* qkvwh  = (half_t*)take(sizeof(half_t) * (size_t)CQ * 3 * CQ);
  half_t* projwh = (half_t*)take(sizeof(half_t) * (size_t)CQ * CQ);
  half_t* fc1wh  = (half_t*)take(sizeof(half_t) * (size_t)CQ * 4 * CQ);
  half_t* fc2wh  = (half_t*)take(sizeof(half_t) * (size_t)4 * CQ * CQ);
  half_t* qh     = (half_t*)take(sizeof(half_t) * (size_t)BQ * HQ * NQ * DHQ);
  half_t* kh     = (half_t*)take(sizeof(half_t) * (size_t)BQ * HQ * KVPAD * DHQ);
  half_t* vh     = (half_t*)take(sizeof(half_t) * (size_t)BQ * HQ * KVPAD * DHQ);
  half_t* attn   = (half_t*)take(sizeof(half_t) * (size_t)BQ * HQ * ATTNR * NPADQ);
  float*  vn2    = (float*) take(sizeof(float)  * (size_t)BQ * NQ);
  float*  sigraw = (float*) take(sizeof(float)  * (size_t)BQ * NQ);
  float*  sig    = (float*) take(sizeof(float)  * (size_t)BQ * NM1Q);
  int*    rowsrc = (int*)   take(sizeof(int)    * (size_t)BQ * NTQ);
  half_t* outh   = (half_t*)take(sizeof(half_t) * (size_t)BQ * NTQ * CQ);
  half_t* x2nh   = (half_t*)take(sizeof(half_t) * (size_t)BQ * NTQ * CQ);
  half_t* hh     = (half_t*)take(sizeof(half_t) * (size_t)BQ * NTQ * 4 * CQ);

  dim3 blk(128);

  zero_kernel<<<64, 256, 0, stream>>>(vn2, (long)BQ * NQ);
  zero_kernel<<<64, 256, 0, stream>>>(sigraw, (long)BQ * NQ);
  // zero padded K/V (rows >= NQ must be zero so padded GEMM cols/K contribute 0)
  zero_half_kernel<<<512, 256, 0, stream>>>(kh, (long)BQ * HQ * KVPAD * DHQ);
  zero_half_kernel<<<512, 256, 0, stream>>>(vh, (long)BQ * HQ * KVPAD * DHQ);
  zero_attn_dummy_kernel<<<BQ * HQ, 256, 0, stream>>>(attn);

  cvt_kernel<<<512, 256, 0, stream>>>(qkvw,  qkvwh,  (long)CQ * 3 * CQ);
  cvt_kernel<<<512, 256, 0, stream>>>(projw, projwh, (long)CQ * CQ);
  cvt_kernel<<<512, 256, 0, stream>>>(fc1w,  fc1wh,  (long)CQ * 4 * CQ);
  cvt_kernel<<<512, 256, 0, stream>>>(fc2w,  fc2wh,  (long)4 * CQ * CQ);

  ln_kernel<<<BQ * NQ, 256, 0, stream>>>(x, n1w, n1b, xnh);

  { // QKV GEMM: per batch, M=785 K=768 Nn=2304
    GemmP p{};
    p.A = xnh; p.aStride = (long)NQ * CQ; p.lda = CQ;
    p.W = qkvwh; p.wStride = 0; p.ldw = 3 * CQ; p.wtrans = 0;
    p.M = NQ; p.Nn = 3 * CQ; p.K = CQ;
    p.rowScale = polin; p.vnorm2 = vn2; p.qh = qh; p.kh = kh; p.vh = vh;
    gemm_kernel<1><<<dim3(13, 36, BQ), blk, 0, stream>>>(p);
  }
  { // scores: per (b,h), S = q k^T * dh^-0.5 -> f16; Nn=800 so pad cols get 0
    GemmP p{};
    p.A = qh; p.aStride = (long)NQ * DHQ; p.lda = DHQ;
    p.W = kh; p.wStride = (long)KVPAD * DHQ; p.ldw = DHQ; p.wtrans = 1;
    p.M = NQ; p.Nn = NPADQ; p.K = DHQ;
    p.outH = attn; p.oHStride = (long)ATTNR * NPADQ; p.oHld = NPADQ; p.alpha = 0.125f;
    gemm_kernel<2><<<dim3(13, 13, BQ * HQ), blk, 0, stream>>>(p);
  }
  softmax_kernel<<<BQ * HQ * NQ, 256, 0, stream>>>(attn, polin, sigraw);
  sig_kernel<<<BQ, 1024, 0, stream>>>(sigraw, vn2, sig);
  select_kernel<<<BQ, 1024, 0, stream>>>(sig, rowsrc, pol);
  selx_kernel<<<BQ * NTQ, 256, 0, stream>>>(x, rowsrc, x2);

  { // AV: gathered attn rows (rowsrc indirection, zrow dummy) x V, K padded to 800
    GemmP p{};
    p.A = attn; p.aStride = (long)ATTNR * NPADQ; p.lda = NPADQ;
    p.amap = rowsrc; p.amapDiv = HQ; p.zrow = NQ;
    p.W = vh; p.wStride = (long)KVPAD * DHQ; p.ldw = DHQ; p.wtrans = 0;
    p.M = NTQ; p.Nn = DHQ; p.K = NPADQ;
    p.outH = outh;
    gemm_kernel<3><<<dim3(7, 1, BQ * HQ), blk, 0, stream>>>(p);
  }
  { // proj: x2 += (out @ proj_w + b) * pol_new
    GemmP p{};
    p.A = outh; p.aStride = (long)NTQ * CQ; p.lda = CQ;
    p.W = projwh; p.wStride = 0; p.ldw = CQ; p.wtrans = 0;
    p.M = NTQ; p.Nn = CQ; p.K = CQ;
    p.outF = x2; p.bias = projb; p.rowScale = pol;
    gemm_kernel<4><<<dim3(7, 12, BQ), blk, 0, stream>>>(p);
  }
  ln_kernel<<<BQ * NTQ, 256, 0, stream>>>(x2, n2w, n2b, x2nh);
  { // fc1 + gelu
    GemmP p{};
    p.A = x2nh; p.aStride = (long)NTQ * CQ; p.lda = CQ;
    p.W = fc1wh; p.wStride = 0; p.ldw = 4 * CQ; p.wtrans = 0;
    p.M = NTQ; p.Nn = 4 * CQ; p.K = CQ;
    p.outH = hh; p.bias = fc1b;
    gemm_kernel<5><<<dim3(7, 48, BQ), blk, 0, stream>>>(p);
  }
  { // fc2: x2 += (h @ fc2_w + b) * pol_new
    GemmP p{};
    p.A = hh; p.aStride = (long)NTQ * 4 * CQ; p.lda = 4 * CQ;
    p.W = fc2wh; p.wStride = 0; p.ldw = CQ; p.wtrans = 0;
    p.M = NTQ; p.Nn = CQ; p.K = 4 * CQ;
    p.outF = x2; p.bias = fc2b; p.rowScale = pol;
    gemm_kernel<6><<<dim3(7, 12, BQ), blk, 0, stream>>>(p);
  }
}
